// MultiHeadAttention_66468913873621
// MI455X (gfx1250) — matrix operand order repaired and checked by reading, not compiled, not run on hardware
//
#include <hip/hip_runtime.h>
#include <hip/hip_bf16.h>

// MI455X / gfx1250: wave32, WMMA bf16 path + async global->LDS staging.
// MHA: B=2, S=2048, DIM=1024, H=16, HD=64. All GEMMs via v_wmma_f32_16x16x32_bf16.

typedef __attribute__((ext_vector_type(16))) __bf16 v16bf;
typedef __attribute__((ext_vector_type(8)))  __bf16 v8bf;
typedef __attribute__((ext_vector_type(8)))  float  v8f;

#define B_   2
#define S_   2048
#define DIM_ 1024
#define H_   16
#define HD_  64

__device__ __forceinline__ unsigned short f2bf(float f) {
    unsigned int u = __builtin_bit_cast(unsigned int, f);
    u += 0x7FFFu + ((u >> 16) & 1u);           // round-to-nearest-even
    return (unsigned short)(u >> 16);
}

// A-fragment (16x32 bf16, M x K). row ptr = &A[row][0] (row = lane&15).
// lanes 0-15: K {0..7, 16..23}; lanes 16-31: K {8..15, 24..31}.
__device__ __forceinline__ v16bf load_a(const unsigned short* row, int kk, int half) {
    v8bf lo = *(const v8bf*)(row + kk +      half * 8);
    v8bf hi = *(const v8bf*)(row + kk + 16 + half * 8);
    return __builtin_shufflevector(lo, hi, 0,1,2,3,4,5,6,7,8,9,10,11,12,13,14,15);
}

// B-fragment (32x16 bf16, K x N), column-major source (col stride = row length).
// lane holds col = lane&15, K {half*8 + 0..7, 16 + half*8 + 0..7} (matches load_a).
__device__ __forceinline__ v16bf load_b(const unsigned short* colrow, int kk, int half) {
    v8bf lo = *(const v8bf*)(colrow + kk +      half * 8);
    v8bf hi = *(const v8bf*)(colrow + kk + 16 + half * 8);
    return __builtin_shufflevector(lo, hi, 0,1,2,3,4,5,6,7,8,9,10,11,12,13,14,15);
}

__device__ __forceinline__ v8f wmma_bf16(v16bf a, v16bf b, v8f c) {
    return __builtin_amdgcn_wmma_f32_16x16x32_bf16(false, a, false, b, (short)0, c, false, false);
}

// gfx1250 async global->LDS: 16B per lane, tracked by ASYNCcnt (not LOADcnt).
__device__ __forceinline__ void async_copy16(unsigned short* ldsDst, const unsigned short* gSrc) {
    unsigned ldsOff = (unsigned)(size_t)ldsDst;            // low 32 bits of generic = LDS offset
    unsigned long long ga = (unsigned long long)gSrc;
    asm volatile("global_load_async_to_lds_b128 %0, %1, off"
                 :: "v"(ldsOff), "v"(ga)
                 : "memory");
}
__device__ __forceinline__ void async_wait0() {
    asm volatile("s_wait_asynccnt 0" ::: "memory");
}

// ---------------- conversion kernels ----------------
__global__ void k_cvt(const float* __restrict__ in, unsigned short* __restrict__ out, int n) {
    int i = blockIdx.x * blockDim.x + threadIdx.x;
    int stride = gridDim.x * blockDim.x;
    for (; i < n; i += stride) out[i] = f2bf(in[i]);
}

// out[b][c][r] = bf16(in[b][r][c])   (transpose within each batch slice)
__global__ void k_cvt_t(const float* __restrict__ in, unsigned short* __restrict__ out,
                        int batch, int rows, int cols) {
    long n = (long)batch * rows * cols;
    long i = (long)blockIdx.x * blockDim.x + threadIdx.x;
    long stride = (long)gridDim.x * blockDim.x;
    for (; i < n; i += stride) {
        int c = (int)(i % cols);
        long t = i / cols;
        int r = (int)(t % rows);
        int b = (int)(t / rows);
        out[((long)b * cols + c) * rows + r] = f2bf(in[i]);
    }
}

// ---------------- fused QKV projection ----------------
// grid.x = B*H, grid.y = S/16. 4 waves/block, wave w owns cols [w*16, w*16+16).
// A tile (16x1024 bf16 = 32KB) staged once per block via async->LDS, shared by all waves.
__global__ __launch_bounds__(128) void k_qkv(
    const unsigned short* __restrict__ xb,
    const unsigned short* __restrict__ wqT, const unsigned short* __restrict__ wkT,
    const unsigned short* __restrict__ wvT,
    const float* __restrict__ bq, const float* __restrict__ bk, const float* __restrict__ bv,
    unsigned short* __restrict__ qO, unsigned short* __restrict__ kO,
    unsigned short* __restrict__ vT)
{
    __shared__ unsigned short xt[16 * DIM_];               // 32 KB A tile
    int bh = blockIdx.x;
    int b = bh / H_, h = bh % H_;
    int s0 = blockIdx.y * 16;
    int tid = threadIdx.x;
    int w = tid >> 5;
    int lane = tid & 31;
    int col = lane & 15, half = lane >> 4;
    int e = w * 16 + col;

    // stage contiguous 16-row x tile into LDS (ASYNCcnt path)
    {
        const unsigned short* src = xb + (long)(b * S_ + s0) * DIM_;
        #pragma unroll
        for (int c = tid; c < 16 * DIM_ / 8; c += 128)
            async_copy16(&xt[c * 8], src + c * 8);
        async_wait0();
    }
    __syncthreads();

    const unsigned short* aRow  = &xt[col * DIM_];
    const unsigned short* bqRow = wqT + ((long)h * HD_ + e) * DIM_;
    const unsigned short* bkRow = wkT + ((long)h * HD_ + e) * DIM_;
    const unsigned short* bvRow = wvT + ((long)h * HD_ + e) * DIM_;

    v8f cq = {}, ck = {}, cv = {};
    // register double-buffer of B frags: next K-step's loads issue before this step's WMMAs
    v16bf bqC = load_b(bqRow, 0, half);
    v16bf bkC = load_b(bkRow, 0, half);
    v16bf bvC = load_b(bvRow, 0, half);
    #pragma unroll 4
    for (int kk = 0; kk < DIM_; kk += 32) {
        int kn = (kk + 32 < DIM_) ? kk + 32 : 0;
        v16bf bqN = load_b(bqRow, kn, half);
        v16bf bkN = load_b(bkRow, kn, half);
        v16bf bvN = load_b(bvRow, kn, half);
        v16bf a = load_a(aRow, kk, half);                  // ds_load_b128 x2
        cq = wmma_bf16(a, bqC, cq);
        ck = wmma_bf16(a, bkC, ck);
        cv = wmma_bf16(a, bvC, cv);
        bqC = bqN; bkC = bkN; bvC = bvN;
    }
    float fq = bq[h * HD_ + e], fk = bk[h * HD_ + e], fv = bv[h * HD_ + e];
    long qkBase = (long)bh * S_ + s0;
    #pragma unroll
    for (int r = 0; r < 8; ++r) {
        int M = r + half * 8;
        qO[(qkBase + M) * HD_ + e] = f2bf(cq[r] + fq);
        kO[(qkBase + M) * HD_ + e] = f2bf(ck[r] + fk);
    }
    // v stored transposed per head: vT[bh][e][s] (feeds B-frags contiguously)
    unsigned short* vp = vT + ((long)bh * HD_ + e) * S_ + s0 + half * 8;
    #pragma unroll
    for (int r = 0; r < 8; ++r) vp[r] = f2bf(cv[r] + fv);
}

// ---------------- flash causal attention ----------------
// grid.x = B*H, grid.y = S/64. Each wave owns one 16-row query tile.
__global__ __launch_bounds__(128) void k_attn(
    const unsigned short* __restrict__ qB,
    const unsigned short* __restrict__ kB,
    const unsigned short* __restrict__ vT,
    unsigned short* __restrict__ ctx)
{
    __shared__ unsigned short pls[4][16 * 32];   // per-wave P tile (16x32 bf16)
    int bh = blockIdx.x;
    int b = bh / H_, h = bh % H_;
    int w = threadIdx.x >> 5;
    int lane = threadIdx.x & 31;
    int col = lane & 15, half = lane >> 4;
    int qTile = blockIdx.y * 4 + w;
    int s0 = qTile * 16;

    const unsigned short* qRow = qB + ((long)bh * S_ + s0 + col) * HD_;
    v16bf aq0 = load_a(qRow, 0, half);
    v16bf aq1 = load_a(qRow, 32, half);

    float m[8], l[8];
    v8f acc[4];
    v8f zero = {};
    #pragma unroll
    for (int r = 0; r < 8; ++r) { m[r] = -3.0e38f; l[r] = 0.0f; }
    #pragma unroll
    for (int nt = 0; nt < 4; ++nt) acc[nt] = zero;

    int tMax = s0 + 15;
    unsigned short* myLds = &pls[w][0];

    for (int tBase = 0; tBase <= tMax; tBase += 32) {
        // load ALL B frags (K and V) up front: V latency hides under softmax VALU work
        const unsigned short* kRow0 = kB + ((long)bh * S_ + tBase + col) * HD_;
        const unsigned short* kRow1 = kRow0 + 16 * HD_;
        v16bf bk00 = load_b(kRow0, 0,  half);
        v16bf bk01 = load_b(kRow0, 32, half);
        v16bf bk10 = load_b(kRow1, 0,  half);
        v16bf bk11 = load_b(kRow1, 32, half);
        const unsigned short* vBase = vT + ((long)bh * HD_ + col) * S_ + tBase;
        v16bf bv0 = load_b(vBase,            0, half);
        v16bf bv1 = load_b(vBase + 16 * S_,  0, half);
        v16bf bv2 = load_b(vBase + 32 * S_,  0, half);
        v16bf bv3 = load_b(vBase + 48 * S_,  0, half);

        // scores: Q (16xHD) x K^T (HD x 32), two 16-wide N tiles, K split 2x32
        v8f sc0 = zero, sc1 = zero;
        sc0 = wmma_bf16(aq0, bk00, sc0);
        sc0 = wmma_bf16(aq1, bk01, sc0);
        sc1 = wmma_bf16(aq0, bk10, sc1);
        sc1 = wmma_bf16(aq1, bk11, sc1);

        // scale + causal mask + streaming softmax stats (rows live in 16-lane halves)
        float p0[8], p1[8], fr[8];
        #pragma unroll
        for (int r = 0; r < 8; ++r) {
            int sQ = s0 + r + half * 8;
            int t0 = tBase + col, t1 = tBase + 16 + col;
            float v0 = sc0[r] * 0.125f; if (t0 > sQ) v0 = -3.0e38f;
            float v1 = sc1[r] * 0.125f; if (t1 > sQ) v1 = -3.0e38f;
            float tm = fmaxf(v0, v1);
            tm = fmaxf(tm, __shfl_xor(tm, 1, 32));
            tm = fmaxf(tm, __shfl_xor(tm, 2, 32));
            tm = fmaxf(tm, __shfl_xor(tm, 4, 32));
            tm = fmaxf(tm, __shfl_xor(tm, 8, 32));
            float mnew = fmaxf(m[r], tm);
            fr[r] = __expf(m[r] - mnew);
            float e0 = __expf(v0 - mnew);
            float e1 = __expf(v1 - mnew);
            p0[r] = e0; p1[r] = e1;
            float rs = e0 + e1;
            rs += __shfl_xor(rs, 1, 32);
            rs += __shfl_xor(rs, 2, 32);
            rs += __shfl_xor(rs, 4, 32);
            rs += __shfl_xor(rs, 8, 32);
            l[r] = l[r] * fr[r] + rs;
            m[r] = mnew;
        }
        #pragma unroll
        for (int nt = 0; nt < 4; ++nt)
            #pragma unroll
            for (int r = 0; r < 8; ++r) acc[nt][r] *= fr[r];

        // C-layout -> A-layout via per-wave LDS bounce
        #pragma unroll
        for (int r = 0; r < 8; ++r) {
            int M = r + half * 8;
            myLds[M * 32 + col]      = f2bf(p0[r]);
            myLds[M * 32 + 16 + col] = f2bf(p1[r]);
        }
        asm volatile("s_wait_dscnt 0" ::: "memory");   // same-wave LDS RAW fence (split counters)
        v16bf ap = load_a(myLds + col * 32, 0, half);

        // P (16x32) x V (32xHD): 4 N tiles over head dim
        acc[0] = wmma_bf16(ap, bv0, acc[0]);
        acc[1] = wmma_bf16(ap, bv1, acc[1]);
        acc[2] = wmma_bf16(ap, bv2, acc[2]);
        acc[3] = wmma_bf16(ap, bv3, acc[3]);
    }
    // normalize and write concatenated-head context (bf16, row-major [B*S][DIM])
    #pragma unroll
    for (int nt = 0; nt < 4; ++nt)
        #pragma unroll
        for (int r = 0; r < 8; ++r) {
            int M = r + half * 8;
            float o = acc[nt][r] / l[r];
            ctx[((long)(b * S_ + s0 + M)) * DIM_ + h * HD_ + nt * 16 + col] = f2bf(o);
        }
}

// ---------------- output projection ----------------
// grid.x = B*S/16 row tiles, grid.y = DIM/64 col groups; 4 waves/block.
// ctx A tile staged via async->LDS, B frags register double-buffered.
__global__ __launch_bounds__(128) void k_proj(
    const unsigned short* __restrict__ ctx,
    const unsigned short* __restrict__ woT,
    const float* __restrict__ bo,
    float* __restrict__ out)
{
    __shared__ unsigned short at[16 * DIM_];               // 32 KB A tile
    int rowTile = blockIdx.x;
    int tid = threadIdx.x;
    int w = tid >> 5;
    int lane = tid & 31;
    int col = lane & 15, half = lane >> 4;
    int n = blockIdx.y * 64 + w * 16 + col;

    {
        const unsigned short* src = ctx + (long)rowTile * 16 * DIM_;
        #pragma unroll
        for (int c = tid; c < 16 * DIM_ / 8; c += 128)
            async_copy16(&at[c * 8], src + c * 8);
        async_wait0();
    }
    __syncthreads();

    const unsigned short* aRow = &at[col * DIM_];
    const unsigned short* bRow = woT + (long)n * DIM_;
    v8f c = {};
    v16bf bC = load_b(bRow, 0, half);
    #pragma unroll 4
    for (int kk = 0; kk < DIM_; kk += 32) {
        int kn = (kk + 32 < DIM_) ? kk + 32 : 0;
        v16bf bN = load_b(bRow, kn, half);
        c = wmma_bf16(load_a(aRow, kk, half), bC, c);
        bC = bN;
    }
    float bias = bo[n];
    #pragma unroll
    for (int r = 0; r < 8; ++r) {
        int M = r + half * 8;
        out[((long)rowTile * 16 + M) * DIM_ + n] = c[r] + bias;
    }
}

extern "C" void kernel_launch(void* const* d_in, const int* in_sizes, int n_in,
                              void* d_out, int out_size, void* d_ws, size_t ws_size,
                              hipStream_t stream) {
    (void)in_sizes; (void)n_in; (void)out_size; (void)ws_size;
    const float* x  = (const float*)d_in[0];
    const float* Wq = (const float*)d_in[1];
    const float* bq = (const float*)d_in[2];
    const float* Wk = (const float*)d_in[3];
    const float* bk = (const float*)d_in[4];
    const float* Wv = (const float*)d_in[5];
    const float* bv = (const float*)d_in[6];
    const float* Wo = (const float*)d_in[7];
    const float* bo = (const float*)d_in[8];
    float* out = (float*)d_out;

    char* ws = (char*)d_ws;                                    // offsets (bytes):
    unsigned short* xb  = (unsigned short*)(ws);               // x bf16       [B*S][DIM]   8 MB
    unsigned short* wqT = (unsigned short*)(ws + (8u  << 20)); // Wq^T bf16    [H][HD][DIM] 2 MB
    unsigned short* wkT = (unsigned short*)(ws + (10u << 20));
    unsigned short* wvT = (unsigned short*)(ws + (12u << 20));
    unsigned short* woT = (unsigned short*)(ws + (14u << 20)); // Wo^T bf16    [DIM][DIM]   2 MB
    unsigned short* qB  = (unsigned short*)(ws + (16u << 20)); // q bf16       [B*H][S][HD] 8 MB
    unsigned short* kB  = (unsigned short*)(ws + (24u << 20)); // k bf16       [B*H][S][HD] 8 MB
    unsigned short* vT  = (unsigned short*)(ws + (32u << 20)); // v^T bf16     [B*H][HD][S] 8 MB
    unsigned short* ctx = (unsigned short*)(ws + (40u << 20)); // ctx bf16     [B*S][DIM]   8 MB

    k_cvt  <<<2048, 256, 0, stream>>>(x,  xb,  B_ * S_ * DIM_);
    k_cvt_t<<<2048, 256, 0, stream>>>(Wq, wqT, H_, DIM_, HD_);
    k_cvt_t<<<2048, 256, 0, stream>>>(Wk, wkT, H_, DIM_, HD_);
    k_cvt_t<<<2048, 256, 0, stream>>>(Wv, wvT, H_, DIM_, HD_);
    k_cvt_t<<<2048, 256, 0, stream>>>(Wo, woT, 1, DIM_, DIM_);

    k_qkv <<<dim3(B_ * H_, S_ / 16), 128, 0, stream>>>(xb, wqT, wkT, wvT, bq, bk, bv, qB, kB, vT);
    k_attn<<<dim3(B_ * H_, S_ / 64), 128, 0, stream>>>(qB, kB, vT, ctx);
    k_proj<<<dim3(B_ * S_ / 16, DIM_ / 64), 128, 0, stream>>>(ctx, woT, bo, out);
}
